// FPBatchNormedLinear_57578331570847
// MI455X (gfx1250) — compile-verified, matrix operand order checked by reading, not run on hardware
//
#include <hip/hip_runtime.h>
#include <hip/hip_fp16.h>

typedef __attribute__((ext_vector_type(16))) _Float16 v16h;
typedef __attribute__((ext_vector_type(8)))  _Float16 v8h;
typedef __attribute__((ext_vector_type(4)))  _Float16 v4h;
typedef __attribute__((ext_vector_type(8)))  float    v8f;
typedef __attribute__((ext_vector_type(4)))  float    v4f;
typedef __attribute__((ext_vector_type(4)))  int      v4i;

#define FP_MINV (-128.0f)
#define FP_MAXV (127.99609375f)
#define FP_S    (256.0f)
#define FP_IS   (0.00390625f)
#define BN_EPS  (1e-5f)

__device__ __forceinline__ float fpq(float x) {
    x = fminf(fmaxf(x, FP_MINV), FP_MAXV);
    return rintf(x * FP_S) * FP_IS;   // round-half-to-even, matches jnp.round
}

// ---- CDNA5 async global->LDS (ASYNCcnt path), builtin if present else asm ----
__device__ __forceinline__ void async_ld_b128(const void* g, void* l) {
#if __has_builtin(__builtin_amdgcn_global_load_async_to_lds_b128)
    __builtin_amdgcn_global_load_async_to_lds_b128(
        (__attribute__((address_space(1))) v4i*)g,
        (__attribute__((address_space(3))) v4i*)l, 0, 0);
#else
    unsigned lofs = (unsigned)(uintptr_t)(__attribute__((address_space(3))) void*)l;
    asm volatile("global_load_async_to_lds_b128 %0, %1, off"
                 :: "v"(lofs), "v"((unsigned long long)(uintptr_t)g)
                 : "memory");
#endif
}

__device__ __forceinline__ void wait_async0() {
#if __has_builtin(__builtin_amdgcn_s_wait_asynccnt)
    __builtin_amdgcn_s_wait_asynccnt(0);
#else
    asm volatile("s_wait_asynccnt 0x0" ::: "memory");
#endif
}

// ---------- prep: x -> f16 ----------
__global__ void cvt_x_f16(const float* __restrict__ x, _Float16* __restrict__ xh, size_t n) {
    size_t i = ((size_t)blockIdx.x * blockDim.x + threadIdx.x) * 4;
    if (i < n) {
        v4f v = *(const v4f*)(x + i);
        v4h h;
        h.x = (_Float16)v.x; h.y = (_Float16)v.y; h.z = (_Float16)v.z; h.w = (_Float16)v.w;
        *(v4h*)(xh + i) = h;
    }
}

// ---------- prep: quantize w -> f16 (quantized values exact in f16) ----------
__global__ void quant_w_f16(const float* __restrict__ w, _Float16* __restrict__ wh, size_t n) {
    size_t i = ((size_t)blockIdx.x * blockDim.x + threadIdx.x) * 4;
    if (i < n) {
        v4f v = *(const v4f*)(w + i);
        v4h h;
        h.x = (_Float16)fpq(v.x); h.y = (_Float16)fpq(v.y);
        h.z = (_Float16)fpq(v.z); h.w = (_Float16)fpq(v.w);
        *(v4h*)(wh + i) = h;
    }
}

// ---------- GEMM + quantize + bias + BN-stats epilogue ----------
#define BM 128
#define BN 128
#define BK 32
#define LDAP 40   // padded LDS row stride (halfs): 80B -> conflict-free b128 frag reads

__launch_bounds__(256)
__global__ void gemm_q_stats(const _Float16* __restrict__ Xh,  // [M,K] row-major
                             const _Float16* __restrict__ Wh,  // [N,K] row-major (B^T layout)
                             const float*    __restrict__ bias,
                             float*          __restrict__ Y,   // [M,N]
                             float*          __restrict__ gSum,
                             float*          __restrict__ gSsq,
                             int M, int N, int K)
{
    __shared__ _Float16 As[2][BM * LDAP];   // double-buffered
    __shared__ _Float16 Bs[2][BN * LDAP];
    __shared__ float colSum[BN];
    __shared__ float colSsq[BN];

    const int tid   = threadIdx.x;
    const int wave  = tid >> 5;       // 8 waves
    const int lane  = tid & 31;
    const int waveM = wave & 1;       // 2 waves along M  (64 rows each)
    const int waveN = wave >> 1;      // 4 waves along N  (32 cols each)
    const int half  = lane >> 4;      // 0/1 lane group
    const int mm    = lane & 15;

    const int mBlock = blockIdx.y * BM;
    const int nBlock = blockIdx.x * BN;

    if (tid < BN) { colSum[tid] = 0.0f; colSsq[tid] = 0.0f; }

    // staging coordinates for this thread (8 halfs = 16B per pass, 2 passes)
    const int r0 = (tid * 8) >> 5;            // 0..63
    const int c0 = (tid * 8) & 31;            // multiple of 8
    const int r1 = r0 + 64;                   // pass 1: rows 64..127

    v8f acc[4][2] = {};

    const int iters = K / BK;

    // kick off tile 0 into buffer 0 (memory -> LDS, no VGPR round-trip)
    async_ld_b128(Xh + (size_t)(mBlock + r0) * K + c0, &As[0][r0 * LDAP + c0]);
    async_ld_b128(Wh + (size_t)(nBlock + r0) * K + c0, &Bs[0][r0 * LDAP + c0]);
    async_ld_b128(Xh + (size_t)(mBlock + r1) * K + c0, &As[0][r1 * LDAP + c0]);
    async_ld_b128(Wh + (size_t)(nBlock + r1) * K + c0, &Bs[0][r1 * LDAP + c0]);

    for (int it = 0; it < iters; ++it) {
        const int cur = it & 1;
        wait_async0();        // this wave's async loads for tile `it` landed
        __syncthreads();      // everyone's landed; prior reads of buf cur^1 done

        if (it + 1 < iters) { // prefetch next tile into other buffer, overlapped with WMMAs
            const int k1 = (it + 1) * BK;
            const int nb = cur ^ 1;
            async_ld_b128(Xh + (size_t)(mBlock + r0) * K + k1 + c0, &As[nb][r0 * LDAP + c0]);
            async_ld_b128(Wh + (size_t)(nBlock + r0) * K + k1 + c0, &Bs[nb][r0 * LDAP + c0]);
            async_ld_b128(Xh + (size_t)(mBlock + r1) * K + k1 + c0, &As[nb][r1 * LDAP + c0]);
            async_ld_b128(Wh + (size_t)(nBlock + r1) * K + k1 + c0, &Bs[nb][r1 * LDAP + c0]);
        }

        // fragments: lane = row mm, two 16B chunks at K=half*8 and K=16+half*8
        v16h a[4], b[2];
#pragma unroll
        for (int i = 0; i < 4; ++i) {
            const _Float16* base = &As[cur][(waveM * 64 + i * 16 + mm) * LDAP + half * 8];
            union { v16h v; v8h h[2]; } u;
            u.h[0] = *(const v8h*)(base);
            u.h[1] = *(const v8h*)(base + 16);
            a[i] = u.v;
        }
#pragma unroll
        for (int j = 0; j < 2; ++j) {
            const _Float16* base = &Bs[cur][(waveN * 32 + j * 16 + mm) * LDAP + half * 8];
            union { v16h v; v8h h[2]; } u;
            u.h[0] = *(const v8h*)(base);
            u.h[1] = *(const v8h*)(base + 16);
            b[j] = u.v;
        }
#pragma unroll
        for (int i = 0; i < 4; ++i)
#pragma unroll
            for (int j = 0; j < 2; ++j)
                acc[i][j] = __builtin_amdgcn_wmma_f32_16x16x32_f16(
                    false, a[i], false, b[j], (short)0, acc[i][j], false, false);
    }

    // epilogue: y = fpq(fpq(acc) + qb); store; accumulate column sums
#pragma unroll
    for (int i = 0; i < 4; ++i) {
#pragma unroll
        for (int j = 0; j < 2; ++j) {
            const int lcol = waveN * 32 + j * 16 + mm;
            const int col  = nBlock + lcol;
            const float qb = fpq(bias[col]);
            const int rowBase = mBlock + waveM * 64 + i * 16 + half * 8;
            float s = 0.0f, s2 = 0.0f;
#pragma unroll
            for (int r = 0; r < 8; ++r) {
                float y = acc[i][j][r];
                y = fpq(y);
                y = fpq(y + qb);
                Y[(size_t)(rowBase + r) * N + col] = y;
                s  += y;
                s2 += y * y;
            }
            atomicAdd(&colSum[lcol], s);   // ds_add_f32
            atomicAdd(&colSsq[lcol], s2);
        }
    }
    __syncthreads();
    if (tid < BN) {
        atomicAdd(&gSum[nBlock + tid], colSum[tid]);  // global_atomic_add_f32
        atomicAdd(&gSsq[nBlock + tid], colSsq[tid]);
    }
}

// ---------- BN scale/shift from accumulated stats ----------
__global__ void bn_params(const float* __restrict__ gSum, const float* __restrict__ gSsq,
                          const float* __restrict__ gamma, const float* __restrict__ beta,
                          float* __restrict__ scale, float* __restrict__ shift,
                          int N, float invM)
{
    int n = blockIdx.x * blockDim.x + threadIdx.x;
    if (n < N) {
        float mean = gSum[n] * invM;
        float var  = fmaxf(gSsq[n] * invM - mean * mean, 0.0f);
        float sc   = gamma[n] * rsqrtf(var + BN_EPS);
        scale[n] = sc;
        shift[n] = beta[n] - mean * sc;
    }
}

// ---------- BN apply + final quantize (bandwidth-bound, vectorized) ----------
__global__ void bn_apply(const float* __restrict__ Y,
                         const float* __restrict__ scale, const float* __restrict__ shift,
                         float* __restrict__ out, int N, size_t total)
{
    size_t i = ((size_t)blockIdx.x * blockDim.x + threadIdx.x) * 4;
    if (i < total) {
        int n = (int)(i % (size_t)N);   // N % 4 == 0 -> aligned column group
        v4f y  = *(const v4f*)(Y + i);
        v4f sc = *(const v4f*)(scale + n);
        v4f sh = *(const v4f*)(shift + n);
        v4f r;
#pragma unroll
        for (int k = 0; k < 4; ++k) r[k] = fpq(fmaf(sc[k], y[k], sh[k]));
        *(v4f*)(out + i) = r;
    }
}

extern "C" void kernel_launch(void* const* d_in, const int* in_sizes, int n_in,
                              void* d_out, int out_size, void* d_ws, size_t ws_size,
                              hipStream_t stream) {
    const float* x     = (const float*)d_in[0];
    const float* w     = (const float*)d_in[1];
    const float* bias  = (const float*)d_in[2];
    const float* gamma = (const float*)d_in[3];
    const float* beta  = (const float*)d_in[4];
    float* out = (float*)d_out;

    const int OUT = in_sizes[2];               // 2048
    const int IN  = in_sizes[1] / OUT;         // 2048
    const int M   = in_sizes[0] / IN;          // 8192

    char* ws = (char*)d_ws;
    size_t o = 0;
    _Float16* xh = (_Float16*)(ws + o); o += (size_t)M  * IN * sizeof(_Float16);
    _Float16* wh = (_Float16*)(ws + o); o += (size_t)OUT * IN * sizeof(_Float16);
    float* Y     = (float*)(ws + o);    o += (size_t)M * OUT * sizeof(float);
    float* gSum  = (float*)(ws + o);    o += (size_t)OUT * sizeof(float);
    float* gSsq  = (float*)(ws + o);    o += (size_t)OUT * sizeof(float);
    float* scale = (float*)(ws + o);    o += (size_t)OUT * sizeof(float);
    float* shift = (float*)(ws + o);

    (void)hipMemsetAsync(gSum, 0, (size_t)OUT * sizeof(float), stream);
    (void)hipMemsetAsync(gSsq, 0, (size_t)OUT * sizeof(float), stream);

    size_t nx = (size_t)M * IN;
    cvt_x_f16<<<(unsigned)((nx / 4 + 255) / 256), 256, 0, stream>>>(x, xh, nx);
    size_t nw = (size_t)OUT * IN;
    quant_w_f16<<<(unsigned)((nw / 4 + 255) / 256), 256, 0, stream>>>(w, wh, nw);

    dim3 grid(OUT / BN, M / BM);
    gemm_q_stats<<<grid, 256, 0, stream>>>(xh, wh, bias, Y, gSum, gSsq, M, OUT, IN);

    bn_params<<<(OUT + 255) / 256, 256, 0, stream>>>(gSum, gSsq, gamma, beta,
                                                     scale, shift, OUT, 1.0f / (float)M);

    size_t tot = (size_t)M * OUT;
    bn_apply<<<(unsigned)((tot / 4 + 255) / 256), 256, 0, stream>>>(Y, scale, shift, out, OUT, tot);
}
